// SAModule_77713138254055
// MI455X (gfx1250) — matrix operand order repaired and checked by reading, not compiled, not run on hardware
//
#include <hip/hip_runtime.h>
#include <hip/hip_bf16.h>
#include <math.h>

typedef __attribute__((ext_vector_type(16))) _Float16 v16h;
typedef __attribute__((ext_vector_type(8)))  _Float16 v8h;
typedef __attribute__((ext_vector_type(4)))  _Float16 v4h;
typedef __attribute__((ext_vector_type(8)))  float    v8f;

#define PI_F 3.14159265358979323846f

// Combine two 16-byte LDS chunks into one v16h (=> 2x ds_load_b128).
__device__ __forceinline__ v16h ld2x8h(const _Float16* p0, const _Float16* p1) {
  const v8h lo = *(const v8h*)p0;
  const v8h hi = *(const v8h*)p1;
  return __builtin_shufflevector(lo, hi, 0, 1, 2, 3, 4, 5, 6, 7,
                                 8, 9, 10, 11, 12, 13, 14, 15);
}

// A-fragment (16x32 f16) from row-major tile[row][k], ld in f16 elems.
// Lane: m = lane&15, half = lane>>4. Chunks: K = kt*32 + 8*half + {0..7 | 16..23}.
__device__ __forceinline__ v16h a_frag_rm(const _Float16* tile, int ld, int rowbase,
                                          int kt, int lane) {
  const int m    = lane & 15;
  const int half = lane >> 4;
  const _Float16* p = tile + (rowbase + m) * ld + kt * 32 + half * 8;
  return ld2x8h(p, p + 16);
}

// B-fragment (32x16 f16) from TRANSPOSED storage Wt[n][k], ld in f16 elems.
// Lane: n = lane&15, half = lane>>4. Chunks: K = kt*32 + 8*half + {0..7 | 16..23}.
__device__ __forceinline__ v16h b_frag_t(const _Float16* Wt, int ld, int nbase,
                                         int kt, int lane) {
  const int n    = lane & 15;
  const int half = lane >> 4;
  const _Float16* p = Wt + (nbase + n) * ld + kt * 32 + half * 8;
  return ld2x8h(p, p + 16);
}

// Float atomic max via monotonic int mapping (valid with -inf init, mixed signs).
__device__ __forceinline__ void atomic_max_f32(float* addr, float val) {
  const int iv = __float_as_int(val);
  if (iv >= 0) atomicMax((int*)addr, iv);
  else         atomicMin((unsigned int*)addr, (unsigned int)iv);
}

// Wave-private LDS staging needs only a compiler scheduling fence: LDS ops from
// one wave are processed in order by the LDS unit (ISA 7.3), so cross-lane
// data passing within a wave is safe without s_barrier.
__device__ __forceinline__ void wave_fence() { __builtin_amdgcn_wave_barrier(); }

// ---- Kernel 0: init agg to -inf ----
__global__ void init_agg_kernel(float* __restrict__ agg, int n) {
  const int i = blockIdx.x * blockDim.x + threadIdx.x;
  if (i < n) agg[i] = __int_as_float(0xFF800000);
}

// ---- Kernel 1: fused edge pipeline ----
__global__ __launch_bounds__(128) void edge_kernel(
    const float* __restrict__ x, const float* __restrict__ pos,
    const int* __restrict__ idx, const int* __restrict__ row, const int* __restrict__ col,
    const float* __restrict__ lw1, const float* __restrict__ lb1,
    const float* __restrict__ lw2, const float* __restrict__ lb2,
    float* __restrict__ agg, int tiles_per_block)
{
  __shared__ __align__(16) _Float16 sW1t[128 * 136];  // W1^T: [hidden][k_in], 34 KB
  __shared__ __align__(16) _Float16 sW2t[128 * 136];  // W2^T: [n_out][k],    34 KB
  __shared__ __align__(16) float    sB1[128];
  __shared__ __align__(16) float    sB2[128];
  __shared__ __align__(16) _Float16 sIn[4][16 * 136]; // per-wave edge_in [edge][k]
  __shared__ float sPD[4][16][4];
  __shared__ int   sSeg[4][16];
  __shared__ int   sCol[4][16];

  const int t = threadIdx.x;
  const int w = t >> 5;
  const int l = t & 31;
  const int half = l >> 4;

  // one-time transposed weight load: i = k*128 + h
  for (int i = t; i < 128 * 128; i += 128) {
    const int k = i >> 7;
    const int h = i & 127;
    sW1t[h * 136 + k] = (k < 127) ? (_Float16)lw1[i] : (_Float16)0.0f;
    sW2t[h * 136 + k] = (_Float16)lw2[i];
  }
  if (t < 128) { sB1[t] = lb1[t]; sB2[t] = lb2[t]; }
  __syncthreads();   // weights ARE shared across waves -> real barrier (once)

  // GEMM2 bias is tile-invariant: hoist this lane's 8 values into registers.
  float bias2[8];
#pragma unroll
  for (int nt = 0; nt < 8; ++nt) bias2[nt] = sB2[nt * 16 + (l & 15)];

  for (int it = 0; it < tiles_per_block; ++it) {
    const int tile = blockIdx.x + it * gridDim.x;
    const long long ebase = (long long)tile * 64 + (long long)w * 16;

    // phase 1: per-edge scalars (wave-private staging)
    if (l < 16) {
      const long long e = ebase + l;
      const int c  = col[e];
      const int r  = row[e];
      const int pi = idx[r];
      sSeg[w][l] = r;
      sCol[w][l] = c;
#pragma unroll
      for (int k = 0; k < 3; ++k)
        sPD[w][l][k] = pos[(long long)c * 3 + k] - pos[(long long)pi * 3 + k];
    }
    wave_fence();

    // phase 2: build edge_in tile [16 x 128] f16; each lane makes 4 consecutive cols
    {
      const int j0 = l * 4;  // lanes 0-15: x cols 0..63; lanes 16-31: cols 64..127
      for (int r = 0; r < 16; ++r) {
        const int c = sCol[w][r];
        v4h o;
        if (j0 < 64) {
          const float4 xv = *(const float4*)(x + (long long)c * 64 + j0);
          o[0] = (_Float16)xv.x; o[1] = (_Float16)xv.y;
          o[2] = (_Float16)xv.z; o[3] = (_Float16)xv.w;
        } else {
#pragma unroll
          for (int q = 0; q < 4; ++q) {
            const int cc = j0 + q;
            float v;
            if (cc < 67) {
              v = sPD[w][r][cc - 64];
            } else if (cc < 127) {
              const int f     = cc - 67;
              const int coord = f / 20;
              const int rem   = f - coord * 20;
              const int fi    = rem >> 1;
              const float s   = sPD[w][r][coord] * (PI_F * (float)(1 << fi));
              v = (rem & 1) ? __cosf(s) : __sinf(s);
            } else {
              v = 0.0f;   // K pad 127 -> 128
            }
            o[q] = (_Float16)v;
          }
        }
        *(v4h*)(&sIn[w][0] + r * 136 + j0) = o;
      }
    }
    wave_fence();

    // GEMM1 (transposed): D[hidden][edge] = W1^T x edge_in^T; bias+ReLU in regs.
    // D-tile nt (lane: edge=l&15, hidden_local=v+8*half) is exactly the GEMM2
    // A-fragment data: a2[kt] elem j <- nt=2kt, elem j+8 <- nt=2kt+1.
    v16h a2[4];
    {
      v16h bE[4];
#pragma unroll
      for (int kt = 0; kt < 4; ++kt) bE[kt] = b_frag_t(&sIn[w][0], 136, 0, kt, l);
#pragma unroll
      for (int nt = 0; nt < 8; ++nt) {
        v8f acc = {};
#pragma unroll
        for (int kt = 0; kt < 4; ++kt) {
          const v16h aw = a_frag_rm(sW1t, 136, nt * 16, kt, l);
          acc = __builtin_amdgcn_wmma_f32_16x16x32_f16(false, aw, false, bE[kt],
                                                       (short)0, acc, false, false);
        }
        const float4 b0 = *(const float4*)(sB1 + nt * 16 + 8 * half);
        const float4 b1 = *(const float4*)(sB1 + nt * 16 + 8 * half + 4);
        const float bb[8] = {b0.x, b0.y, b0.z, b0.w, b1.x, b1.y, b1.z, b1.w};
#pragma unroll
        for (int v = 0; v < 8; ++v) {
          const float hv = fmaxf(acc[v] + bb[v], 0.0f);
          a2[nt >> 1][(nt & 1) * 8 + v] = (_Float16)hv;
        }
      }
    }

    // GEMM2: edge_feat = h @ W2 + b2 (A in registers), then segmented atomic max
#pragma unroll
    for (int nt = 0; nt < 8; ++nt) {
      v8f acc = {};
#pragma unroll
      for (int kt = 0; kt < 4; ++kt) {
        const v16h bw = b_frag_t(sW2t, 136, nt * 16, kt, l);
        acc = __builtin_amdgcn_wmma_f32_16x16x32_f16(false, a2[kt], false, bw,
                                                     (short)0, acc, false, false);
      }
      const int   ncol = nt * 16 + (l & 15);
      const float bias = bias2[nt];
      int   prev = -1;
      float run  = 0.0f;
#pragma unroll
      for (int v = 0; v < 8; ++v) {
        const int   m   = v + 8 * half;     // consecutive tile rows (row[] sorted)
        const int   seg = sSeg[w][m];
        const float val = acc[v] + bias;
        if (seg != prev) {
          if (prev >= 0) atomic_max_f32(&agg[(long long)prev * 128 + ncol], run);
          prev = seg; run = val;
        } else {
          run = fmaxf(run, val);
        }
      }
      if (prev >= 0) atomic_max_f32(&agg[(long long)prev * 128 + ncol], run);
    }
    wave_fence();   // next iteration's phase-1 writes vs this wave's reads
  }
}

// ---- Kernel 2: out = fixup(agg) @ gw + gb ----
__global__ __launch_bounds__(128) void gemm3_kernel(
    const float* __restrict__ agg, const float* __restrict__ gw,
    const float* __restrict__ gb, float* __restrict__ out)
{
  __shared__ __align__(16) _Float16 sGWt[256 * 136];   // gw^T: [n][k], 68 KB
  __shared__ __align__(16) float    sGB[256];
  __shared__ __align__(16) _Float16 sA[4][16 * 136];

  const int t = threadIdx.x;
  const int w = t >> 5;
  const int l = t & 31;
  const int half = l >> 4;

  for (int i = t; i < 128 * 256; i += 128) {   // i = k*256 + n
    const int k = i >> 8;
    const int n = i & 255;
    sGWt[n * 136 + k] = (_Float16)gw[i];
  }
  for (int i = t; i < 256; i += 128) sGB[i] = gb[i];

  const int rowbase = blockIdx.x * 64 + w * 16;
  for (int r = 0; r < 16; ++r) {
    const long long m = rowbase + r;
    float4 v4 = *(const float4*)(agg + m * 128 + l * 4);
    if (v4.x < -3.0e38f) v4.x = 0.0f;
    if (v4.y < -3.0e38f) v4.y = 0.0f;
    if (v4.z < -3.0e38f) v4.z = 0.0f;
    if (v4.w < -3.0e38f) v4.w = 0.0f;
    v4h o;
    o[0] = (_Float16)v4.x; o[1] = (_Float16)v4.y;
    o[2] = (_Float16)v4.z; o[3] = (_Float16)v4.w;
    *(v4h*)(&sA[w][0] + r * 136 + l * 4) = o;
  }
  __syncthreads();   // sGWt is shared across the 4 waves

  v16h a[4];
#pragma unroll
  for (int kt = 0; kt < 4; ++kt) a[kt] = a_frag_rm(&sA[w][0], 136, 0, kt, l);
#pragma unroll
  for (int nt = 0; nt < 16; ++nt) {
    v8f acc = {};
#pragma unroll
    for (int kt = 0; kt < 4; ++kt) {
      const v16h b = b_frag_t(sGWt, 136, nt * 16, kt, l);
      acc = __builtin_amdgcn_wmma_f32_16x16x32_f16(false, a[kt], false, b,
                                                   (short)0, acc, false, false);
    }
    const int   ncol = nt * 16 + (l & 15);
    const float bias = sGB[ncol];
#pragma unroll
    for (int v = 0; v < 8; ++v) {
      const long long m = rowbase + v + 8 * half;
      out[m * 256 + ncol] = acc[v] + bias;
    }
  }
}

// ---- Kernel 3: pos_idx and batch[idx] outputs ----
__global__ void aux_kernel(const float* __restrict__ pos, const int* __restrict__ batch,
                           const int* __restrict__ idx, float* __restrict__ pos_out,
                           float* __restrict__ batch_out, int M)
{
  const int m = blockIdx.x * blockDim.x + threadIdx.x;
  if (m >= M) return;
  const int pi = idx[m];
  pos_out[(long long)m * 3 + 0] = pos[(long long)pi * 3 + 0];
  pos_out[(long long)m * 3 + 1] = pos[(long long)pi * 3 + 1];
  pos_out[(long long)m * 3 + 2] = pos[(long long)pi * 3 + 2];
  batch_out[m] = (float)batch[pi];
}

extern "C" void kernel_launch(void* const* d_in, const int* in_sizes, int n_in,
                              void* d_out, int out_size, void* d_ws, size_t ws_size,
                              hipStream_t stream) {
  const float* x     = (const float*)d_in[0];
  const float* pos   = (const float*)d_in[1];
  const int*   batch = (const int*)  d_in[2];
  const int*   idx   = (const int*)  d_in[3];
  const int*   row   = (const int*)  d_in[4];
  const int*   col   = (const int*)  d_in[5];
  const float* lw1   = (const float*)d_in[6];
  const float* lb1   = (const float*)d_in[7];
  const float* lw2   = (const float*)d_in[8];
  const float* lb2   = (const float*)d_in[9];
  const float* gw    = (const float*)d_in[10];
  const float* gb    = (const float*)d_in[11];

  const int M = in_sizes[3];            // 16384
  const int E = in_sizes[4];            // 524288

  float* out = (float*)d_out;           // [M*256 | M*3 | M]
  float* agg = (float*)d_ws;            // M*128 floats

  init_agg_kernel<<<(M * 128 + 255) / 256, 256, 0, stream>>>(agg, M * 128);

  const int num_tiles = E / 64;
  int grid = 1024;
  while (grid > 1 && (num_tiles % grid)) grid >>= 1;
  const int tiles_per_block = num_tiles / grid;
  edge_kernel<<<grid, 128, 0, stream>>>(x, pos, idx, row, col,
                                        lw1, lb1, lw2, lb2, agg, tiles_per_block);

  gemm3_kernel<<<M / 64, 128, 0, stream>>>(agg, gw, gb, out);

  aux_kernel<<<(M + 255) / 256, 256, 0, stream>>>(
      pos, batch, idx, out + (long long)M * 256,
      out + (long long)M * 256 + (long long)M * 3, M);
}
